// DecoderWithAttention_74586402063093
// MI455X (gfx1250) — compile-verified, matrix operand order checked by reading, not run on hardware
//
#include <hip/hip_runtime.h>
#include <hip/hip_bf16.h>
#include <math.h>

// Problem constants (from reference)
#define Vv 32000
#define Ee 256
#define Hh 512
#define Aa 64
#define Bb 32
#define Tt 64
#define Ss 128
#define XDIM (Ee + 2 * Hh)   // 1280
#define G3H (3 * Hh)         // 1536

// ---------------- WMMA types ----------------
typedef __attribute__((ext_vector_type(16))) __bf16 v16bf;
typedef __attribute__((ext_vector_type(8)))  float  v8f;

union FragBF {
    uint4  q[2];
    v16bf  v;
};

static __device__ __forceinline__ unsigned short f32_to_bf16(float f) {
    unsigned int u = __float_as_uint(f);
    unsigned int r = (u + 0x7FFFu + ((u >> 16) & 1u)) >> 16;  // RNE
    return (unsigned short)r;
}

static __device__ __forceinline__ float sigmoidf_(float x) {
    return 1.0f / (1.0f + expf(-x));
}

// ---------------- Workspace layout (bytes, 256-aligned) ----------------
#define OFS_FCWBF 0UL          // V*H bf16            = 32,768,000
#define OFS_WIHBF 32768000UL   // 3H*XDIM bf16        =  3,932,160
#define OFS_WHHBF 36700160UL   // 3H*H bf16           =  1,572,864
#define OFS_HBF   38273024UL   // B*T*H bf16          =  2,097,152
#define OFS_WH    40370176UL   // B*S*A f32           =  1,048,576
#define OFS_HBUF  41418752UL   // 2 * B*H f32         =    131,072
#define OFS_HBBUF 41549824UL   // 2 * B*H bf16        =     65,536
#define OFS_XBF   41615360UL   // B*XDIM bf16         =     81,920
#define OFS_GI    41697280UL   // B*3H f32            =    196,608
#define OFS_GH    41893888UL   // B*3H f32            =    196,608

// ================= Prep kernels =================

// generic f32 -> bf16 (grid-stride)
__global__ void k_f32_to_bf16(const float* __restrict__ src,
                              unsigned short* __restrict__ dst, int n) {
    for (int i = blockIdx.x * blockDim.x + threadIdx.x; i < n;
         i += gridDim.x * blockDim.x)
        dst[i] = f32_to_bf16(src[i]);
}

// Wh = encoder_outputs @ W_h : [B,S,A]
__global__ void k_prep_wh(const float* __restrict__ enc,
                          const float* __restrict__ W_h,
                          float* __restrict__ wh) {
    int bs = blockIdx.x;                 // 0 .. B*S-1
    int a  = threadIdx.x;                // 0 .. 63
    const float* e = enc + (size_t)bs * (2 * Hh);
    float acc = 0.0f;
    for (int k = 0; k < 2 * Hh; ++k)
        acc = fmaf(e[k], W_h[k * Aa + a], acc);
    wh[bs * Aa + a] = acc;
}

// h0 = hidden[-1] (fp32 + bf16 copies)
__global__ void k_copy_h0(const float* __restrict__ hidden,
                          float* __restrict__ hbuf0,
                          unsigned short* __restrict__ hbb0) {
    int i = blockIdx.x * blockDim.x + threadIdx.x;
    if (i < Bb * Hh) {
        float v = hidden[i];
        hbuf0[i] = v;
        hbb0[i]  = f32_to_bf16(v);
    }
}

// ================= Per-step kernel 1: attention =================
// one block per batch row, 128 threads
__global__ void k_attn(int t,
                       const int* __restrict__ trg,            // [B,T]
                       const float* __restrict__ enc,          // [B,S,2H]
                       const int* __restrict__ mask,           // [B,S]
                       const float* __restrict__ emb_table,    // [V,E]
                       const float* __restrict__ W_s,          // [H,A]
                       const float* __restrict__ v_a,          // [A]
                       const float* __restrict__ wh,           // [B,S,A]
                       const float* __restrict__ hprev,        // [B,H] fp32
                       unsigned short* __restrict__ xbf,       // [B,XDIM] bf16
                       float* __restrict__ attn_out)           // [B,T,S]
{
    __shared__ float ws_l[Aa];
    __shared__ float attn_l[Ss];
    __shared__ float red[Ss];

    const int b   = blockIdx.x;
    const int tid = threadIdx.x;

    // Ws[b][a] = sum_k h[b][k] * W_s[k][a]
    if (tid < Aa) {
        const float* hb = hprev + b * Hh;
        float acc = 0.0f;
        for (int k = 0; k < Hh; ++k)
            acc = fmaf(hb[k], W_s[k * Aa + tid], acc);
        ws_l[tid] = acc;
    }
    // embedding gather into x[:, 0:E] (bf16)
    {
        int tok = trg[b * Tt + t];
        for (int e = tid; e < Ee; e += blockDim.x)
            xbf[b * XDIM + e] = f32_to_bf16(emb_table[(size_t)tok * Ee + e]);
    }
    __syncthreads();

    // score[s] = v_a . tanh(Wh[b,s,:] + Ws)
    const int s = tid;
    const float* whp = wh + ((size_t)b * Ss + s) * Aa;
    float sc = 0.0f;
    for (int a = 0; a < Aa; ++a)
        sc = fmaf(v_a[a], tanhf(whp[a] + ws_l[a]), sc);
    if (mask[b * Ss + s] == 0) sc = -1e9f;

    // softmax over S=128
    red[tid] = sc;
    __syncthreads();
    for (int off = Ss / 2; off > 0; off >>= 1) {
        if (tid < off) red[tid] = fmaxf(red[tid], red[tid + off]);
        __syncthreads();
    }
    float m = red[0];
    __syncthreads();
    float ex = expf(sc - m);
    red[tid] = ex;
    __syncthreads();
    for (int off = Ss / 2; off > 0; off >>= 1) {
        if (tid < off) red[tid] += red[tid + off];
        __syncthreads();
    }
    float attn = ex / red[0];
    attn_l[tid] = attn;
    attn_out[((size_t)b * Tt + t) * Ss + s] = attn;
    __syncthreads();

    // context[d] = sum_s attn[s] * enc[b,s,d]  -> x[:, E : E+2H] (bf16)
    for (int d = tid; d < 2 * Hh; d += blockDim.x) {
        float c = 0.0f;
        const float* eb = enc + ((size_t)b * Ss) * (2 * Hh) + d;
        for (int si = 0; si < Ss; ++si)
            c = fmaf(attn_l[si], eb[(size_t)si * (2 * Hh)], c);
        xbf[b * XDIM + Ee + d] = f32_to_bf16(c);
    }
}

// ================= Per-step kernel 2: GRU matmuls via WMMA =================
// gi[32,1536] = x[32,1280] @ W_ih^T + b_ih   (blockIdx.y == 0)
// gh[32,1536] = h[32, 512] @ W_hh^T + b_hh   (blockIdx.y == 1)
// grid (12, 2), 256 threads = 8 waves; each wave: 16 N-cols x 32 M-rows (2 tiles)
__global__ void __launch_bounds__(256)
k_gru_mm(const unsigned short* __restrict__ xbf,      // [32,XDIM] bf16
         const unsigned short* __restrict__ hb_prev,  // [32,H] bf16
         const unsigned short* __restrict__ wih_bf,   // [3H,XDIM] bf16
         const unsigned short* __restrict__ whh_bf,   // [3H,H] bf16
         const float* __restrict__ b_ih,
         const float* __restrict__ b_hh,
         float* __restrict__ gibuf,                   // [32,3H]
         float* __restrict__ ghbuf)                   // [32,3H]
{
    const int tid  = threadIdx.x;
    const int wave = tid >> 5;
    const int lane = tid & 31;
    const int lrow = lane & 15;
    const int hi   = lane >> 4;
    const int off0 = hi * 8;
    const int off1 = 16 + hi * 8;

    const bool is_gh = (blockIdx.y != 0);
    const unsigned short* Am = is_gh ? hb_prev : xbf;
    const unsigned short* Bw = is_gh ? whh_bf  : wih_bf;
    const float*          bv = is_gh ? b_hh    : b_ih;
    float*                Cg = is_gh ? ghbuf   : gibuf;
    const int Kdim = is_gh ? Hh : XDIM;

    const int nb  = blockIdx.x * 128 + wave * 16;
    const int col = nb + lrow;

    const unsigned short* bp  = Bw + (size_t)col * Kdim;
    const unsigned short* ap0 = Am + (size_t)lrow * Kdim;
    const unsigned short* ap1 = Am + (size_t)(16 + lrow) * Kdim;

    v8f acc0 = {0.f, 0.f, 0.f, 0.f, 0.f, 0.f, 0.f, 0.f};
    v8f acc1 = {0.f, 0.f, 0.f, 0.f, 0.f, 0.f, 0.f, 0.f};

    for (int k = 0; k < Kdim; k += 32) {
        FragBF bf, a0, a1;
        bf.q[0] = *reinterpret_cast<const uint4*>(bp + k + off0);
        bf.q[1] = *reinterpret_cast<const uint4*>(bp + k + off1);
        a0.q[0] = *reinterpret_cast<const uint4*>(ap0 + k + off0);
        a0.q[1] = *reinterpret_cast<const uint4*>(ap0 + k + off1);
        a1.q[0] = *reinterpret_cast<const uint4*>(ap1 + k + off0);
        a1.q[1] = *reinterpret_cast<const uint4*>(ap1 + k + off1);
        acc0 = __builtin_amdgcn_wmma_f32_16x16x32_bf16(
            false, a0.v, false, bf.v, (short)0, acc0, false, false);
        acc1 = __builtin_amdgcn_wmma_f32_16x16x32_bf16(
            false, a1.v, false, bf.v, (short)0, acc1, false, false);
    }

    const float bias = bv[col];
    const int mbase = hi * 8;
    #pragma unroll
    for (int i = 0; i < 8; ++i) {
        Cg[(size_t)(mbase + i)      * G3H + col] = acc0[i] + bias;
        Cg[(size_t)(mbase + i + 16) * G3H + col] = acc1[i] + bias;
    }
}

// ================= Per-step kernel 3: GRU gate combine =================
__global__ void k_gru_gate(int t,
                           const float* __restrict__ gibuf,   // [32,3H]
                           const float* __restrict__ ghbuf,   // [32,3H]
                           const float* __restrict__ hprev,   // [B,H] fp32
                           float* __restrict__ hnext,         // [B,H] fp32
                           unsigned short* __restrict__ hbnext,  // [B,H] bf16
                           unsigned short* __restrict__ hbf)  // [B*T,H] bf16
{
    int i = blockIdx.x * blockDim.x + threadIdx.x;  // 0 .. B*H-1
    if (i >= Bb * Hh) return;
    int b = i >> 9;         // /512
    int u = i & (Hh - 1);   // %512

    const float* gi = gibuf + (size_t)b * G3H;
    const float* gh = ghbuf + (size_t)b * G3H;
    float r  = sigmoidf_(gi[u]          + gh[u]);
    float z  = sigmoidf_(gi[Hh + u]     + gh[Hh + u]);
    float n  = tanhf(gi[2 * Hh + u] + r * gh[2 * Hh + u]);
    float hp = hprev[i];
    float hnew = (1.0f - z) * n + z * hp;

    hnext[i]  = hnew;
    unsigned short hv = f32_to_bf16(hnew);
    hbnext[i] = hv;
    hbf[((size_t)b * Tt + t) * Hh + u] = hv;
}

// ================= Phase 2: batched logits GEMM via WMMA =================
// C[M=2048, N=32000] = Hbf[M,512] * fcw_bf[N,512]^T + fc_b
// block: 256 threads (8 waves); tile 64(M) x 128(N); A tile (64KB) in LDS.
__global__ void __launch_bounds__(256)
k_logits_wmma(const unsigned short* __restrict__ hbf,
              const unsigned short* __restrict__ fcw_bf,
              const float* __restrict__ fc_b,
              float* __restrict__ out)
{
    __shared__ __align__(16) unsigned short As[64 * Hh];  // 64 KB

    const int tid = threadIdx.x;
    const int M0  = blockIdx.x * 64;
    const int N0  = blockIdx.y * 128;

    // cooperative load of A tile (rows M0..M0+63, contiguous in hbf)
    {
        const uint4* src = reinterpret_cast<const uint4*>(hbf + (size_t)M0 * Hh);
        uint4* dst = reinterpret_cast<uint4*>(As);
        const int nvec = 64 * Hh / 8;  // 4096 uint4
        for (int i = tid; i < nvec; i += 256) dst[i] = src[i];
    }
    __syncthreads();

    const int wave = tid >> 5;
    const int lane = tid & 31;
    const int lrow = lane & 15;
    const int hi   = lane >> 4;
    const int off0 = hi * 8;
    const int off1 = 16 + hi * 8;

    const int nb = N0 + wave * 16;
    const unsigned short* bp = fcw_bf + (size_t)(nb + lrow) * Hh;
    const unsigned short* ap[4];
    #pragma unroll
    for (int j = 0; j < 4; ++j)
        ap[j] = As + (size_t)(j * 16 + lrow) * Hh;

    v8f acc[4];
    #pragma unroll
    for (int j = 0; j < 4; ++j)
        acc[j] = (v8f){0.f, 0.f, 0.f, 0.f, 0.f, 0.f, 0.f, 0.f};

    for (int k = 0; k < Hh; k += 32) {
        FragBF bf;
        bf.q[0] = *reinterpret_cast<const uint4*>(bp + k + off0);
        bf.q[1] = *reinterpret_cast<const uint4*>(bp + k + off1);
        if (k + 32 < Hh)
            __builtin_prefetch(bp + k + 32, 0, 3);
        #pragma unroll
        for (int j = 0; j < 4; ++j) {
            FragBF a;
            a.q[0] = *reinterpret_cast<const uint4*>(ap[j] + k + off0);
            a.q[1] = *reinterpret_cast<const uint4*>(ap[j] + k + off1);
            acc[j] = __builtin_amdgcn_wmma_f32_16x16x32_bf16(
                false, a.v, false, bf.v, (short)0, acc[j], false, false);
        }
    }

    const float bias = fc_b[nb + lrow];
    #pragma unroll
    for (int j = 0; j < 4; ++j) {
        const int mbase = M0 + j * 16 + hi * 8;
        #pragma unroll
        for (int i = 0; i < 8; ++i)
            out[(size_t)(mbase + i) * Vv + nb + lrow] = acc[j][i] + bias;
    }
}

// ================= Launch =================
extern "C" void kernel_launch(void* const* d_in, const int* in_sizes, int n_in,
                              void* d_out, int out_size, void* d_ws, size_t ws_size,
                              hipStream_t stream) {
    const int*   trg    = (const int*)  d_in[0];
    const float* enc    = (const float*)d_in[1];
    const int*   mask   = (const int*)  d_in[2];
    const float* hidden = (const float*)d_in[3];
    const float* emb    = (const float*)d_in[4];
    const float* W_h    = (const float*)d_in[5];
    const float* W_s    = (const float*)d_in[6];
    const float* v_a    = (const float*)d_in[7];
    const float* W_ih   = (const float*)d_in[8];
    const float* W_hh   = (const float*)d_in[9];
    const float* b_ih   = (const float*)d_in[10];
    const float* b_hh   = (const float*)d_in[11];
    const float* fc_w   = (const float*)d_in[12];
    const float* fc_b   = (const float*)d_in[13];

    float* out_logits = (float*)d_out;                          // [B,T,V]
    float* out_attn   = (float*)d_out + (size_t)Bb * Tt * Vv;   // [B,T,S]

    char* ws = (char*)d_ws;
    unsigned short* fcw_bf = (unsigned short*)(ws + OFS_FCWBF);
    unsigned short* wih_bf = (unsigned short*)(ws + OFS_WIHBF);
    unsigned short* whh_bf = (unsigned short*)(ws + OFS_WHHBF);
    unsigned short* hbf    = (unsigned short*)(ws + OFS_HBF);
    float*          wh     = (float*)(ws + OFS_WH);
    float*          hbuf   = (float*)(ws + OFS_HBUF);           // 2 x [B,H]
    unsigned short* hbbuf  = (unsigned short*)(ws + OFS_HBBUF); // 2 x [B,H]
    unsigned short* xbf    = (unsigned short*)(ws + OFS_XBF);
    float*          gibuf  = (float*)(ws + OFS_GI);
    float*          ghbuf  = (float*)(ws + OFS_GH);

    // ---- prep: weight conversions (bf16), Wh, h0 ----
    k_f32_to_bf16<<<4096, 256, 0, stream>>>(fc_w, fcw_bf, Vv * Hh);
    k_f32_to_bf16<<<2048, 256, 0, stream>>>(W_ih, wih_bf, G3H * XDIM);
    k_f32_to_bf16<<<1024, 256, 0, stream>>>(W_hh, whh_bf, G3H * Hh);
    k_prep_wh<<<Bb * Ss, Aa, 0, stream>>>(enc, W_h, wh);
    k_copy_h0<<<(Bb * Hh + 255) / 256, 256, 0, stream>>>(hidden, hbuf, hbbuf);

    // ---- recurrence: 64 sequential steps ----
    for (int t = 0; t < Tt; ++t) {
        const int cur = t & 1, nxt = (t + 1) & 1;
        const float*          hp  = hbuf  + cur * (Bb * Hh);
        float*                hn  = hbuf  + nxt * (Bb * Hh);
        const unsigned short* hbp = hbbuf + cur * (Bb * Hh);
        unsigned short*       hbn = hbbuf + nxt * (Bb * Hh);

        k_attn<<<Bb, Ss, 0, stream>>>(t, trg, enc, mask, emb, W_s, v_a, wh,
                                      hp, xbf, out_attn);
        k_gru_mm<<<dim3(G3H / 128, 2), 256, 0, stream>>>(
            xbf, hbp, wih_bf, whh_bf, b_ih, b_hh, gibuf, ghbuf);
        k_gru_gate<<<(Bb * Hh + 255) / 256, 256, 0, stream>>>(
            t, gibuf, ghbuf, hp, hn, hbn, hbf);
    }

    // ---- batched logits GEMM (WMMA) ----
    {
        dim3 grid((Bb * Tt) / 64, Vv / 128);   // (32, 250)
        k_logits_wmma<<<grid, 256, 0, stream>>>(hbf, fcw_bf, fc_b, out_logits);
    }
}
